// AtlasV4Transformer_77799037599845
// MI455X (gfx1250) — matrix operand order repaired and checked by reading, not compiled, not run on hardware
//
#include <hip/hip_runtime.h>
#include <math.h>

#define B 4
#define G 30
#define D 320
#define L 4
#define NH 8
#define DK 40
#define S 900
#define FF 1280
#define LN_EPS 1e-5f
#define DT_PAD 340   // padded row stride for [xs|tp] concat (337 -> 340, keeps float4 alignment)

typedef __attribute__((ext_vector_type(16))) _Float16 v16h;
typedef __attribute__((ext_vector_type(8)))  _Float16 v8h;
typedef __attribute__((ext_vector_type(4)))  _Float16 v4h;
typedef __attribute__((ext_vector_type(8)))  float    v8f;
typedef __attribute__((ext_vector_type(4)))  float    v4f;

#define BM 128
#define BN 64
#define BK 64
#define LDSA (BK + 8)   // 72 halves per row, 144B (16B multiple)
#define LDSB (BK + 8)

__device__ __forceinline__ float gelu_erf(float x) {
  return 0.5f * x * (1.0f + erff(x * 0.70710678118654752440f));
}

// ---------------------------------------------------------------------------
// Generic batched WMMA GEMM:
//   C[z] = act( alpha * A[z](MxK) * B[z](KxN) + bias1[n] + bias2d[m,n] )
// z offsets: off = (z/zInner)*s? + (z%zInner)*s?2
// AMODE==1: A rows are conv-shifted reads of x[B,S,D] with tap (convDY,convDX)
// accumulate==1: C += alpha * A*B (no bias/act)
// ---------------------------------------------------------------------------
template<int AMODE, int TRANSB>
__launch_bounds__(256)
__global__ void gemm_wmma_kernel(
    const float* __restrict__ A, const float* __restrict__ Bm,
    const float* __restrict__ bias1, const float* __restrict__ bias2d,
    float* __restrict__ C,
    int M, int N, int K, int lda, int ldb, int ldc,
    long sA, long sA2, long sB, long sB2, long sC, long sC2, long sBias2,
    int zInner, float alpha, int act, int accumulate, int convDY, int convDX)
{
  __shared__ _Float16 shA[BM][LDSA];
  __shared__ _Float16 shBt[BN][LDSB];   // transposed: [n][k]

  const int tid = threadIdx.x;
  const long z  = blockIdx.z;
  const long zo = z / zInner;
  const long zi = z - zo * zInner;
  A  += zo * sA + zi * sA2;
  Bm += zo * sB + zi * sB2;
  C  += zo * sC + zi * sC2;
  const float* b2 = bias2d ? (bias2d + zi * sBias2) : (const float*)0;

  const int bm = blockIdx.y * BM;
  const int bn = blockIdx.x * BN;

  const int wid  = tid >> 5;
  const int lane = tid & 31;
  const int wm = wid & 3;    // 4 waves along M, 32 rows each
  const int wn = wid >> 2;   // 2 waves along N, 32 cols each
  const int lr = lane & 15;
  const int lh = lane >> 4;

  const bool ldaVec = ((lda & 3) == 0);
  const bool ldbVec = ((ldb & 3) == 0);

  v8f acc[2][2];
  #pragma unroll
  for (int i = 0; i < 2; i++)
    #pragma unroll
    for (int j = 0; j < 2; j++)
      #pragma unroll
      for (int e = 0; e < 8; e++) acc[i][j][e] = 0.0f;

  for (int kk = 0; kk < K; kk += BK) {
    // ---- stage A tile [BM x BK] as f16 (float4 groups) ----
    #pragma unroll
    for (int t = 0; t < (BM * BK) / (4 * 256); t++) {
      int g = tid + t * 256;
      int r = g >> 4;            // BK/4 = 16 groups per row
      int c = (g & 15) << 2;
      int m = bm + r;
      int k = kk + c;
      long arow = 0; bool mv = false;
      if (AMODE == 0) {
        arow = m; mv = (m < M);
      } else {
        if (m < M) {
          int b_ = m / S; int rem = m - b_ * S;
          int h = rem / G; int w = rem - h * G;
          int sh = h + convDY, sw = w + convDX;
          if (sh >= 0 && sh < G && sw >= 0 && sw < G) {
            mv = true; arow = (long)b_ * S + (long)sh * G + sw;
          }
        }
      }
      v4h hv;
      if (mv && ldaVec && (k + 3 < K)) {
        v4f f = *(const v4f*)(A + arow * lda + k);
        hv[0] = (_Float16)f[0]; hv[1] = (_Float16)f[1];
        hv[2] = (_Float16)f[2]; hv[3] = (_Float16)f[3];
        if (t == 0 && kk + BK < K) __builtin_prefetch(A + arow * lda + k + BK, 0, 1);
      } else {
        long ar = mv ? arow : 0;
        #pragma unroll
        for (int e = 0; e < 4; e++) {
          int ke = k + e;
          int kc = (ke < K) ? ke : (K - 1);
          float f = A[ar * lda + kc];
          hv[e] = (_Float16)((mv && ke < K) ? f : 0.0f);
        }
      }
      *(v4h*)&shA[r][c] = hv;
    }

    // ---- stage B tile into shBt[n][k] ----
    if (TRANSB) {
      #pragma unroll
      for (int t = 0; t < (BN * BK) / (4 * 256); t++) {
        int g = tid + t * 256;
        int r = g >> 4;            // n
        int c = (g & 15) << 2;     // k
        int n = bn + r;
        int k = kk + c;
        v4h hv;
        if ((n < N) && ldbVec && (k + 3 < K)) {
          v4f f = *(const v4f*)(Bm + (long)n * ldb + k);
          hv[0] = (_Float16)f[0]; hv[1] = (_Float16)f[1];
          hv[2] = (_Float16)f[2]; hv[3] = (_Float16)f[3];
          if (t == 0 && kk + BK < K) __builtin_prefetch(Bm + (long)n * ldb + k + BK, 0, 1);
        } else {
          long nr = (n < N) ? n : (N - 1);
          #pragma unroll
          for (int e = 0; e < 4; e++) {
            int ke = k + e;
            int kc = (ke < K) ? ke : (K - 1);
            float f = Bm[nr * ldb + kc];
            hv[e] = (_Float16)(((n < N) && ke < K) ? f : 0.0f);
          }
        }
        *(v4h*)&shBt[r][c] = hv;
      }
    } else {
      #pragma unroll
      for (int t = 0; t < (BN * BK) / (4 * 256); t++) {
        int g = tid + t * 256;
        int r = g >> 4;            // k (BN/4 = 16 groups per k row)
        int c = (g & 15) << 2;     // n
        int k = kk + r;
        int n = bn + c;
        float f0, f1, f2, f3;
        if ((k < K) && ldbVec && (n + 3 < N)) {
          v4f f = *(const v4f*)(Bm + (long)k * ldb + n);
          f0 = f[0]; f1 = f[1]; f2 = f[2]; f3 = f[3];
          if (t == 0 && kk + BK < K) __builtin_prefetch(Bm + (long)(k + BK) * ldb + n, 0, 1);
        } else {
          long kr = (k < K) ? k : (K - 1);
          int n0 = (n + 0 < N) ? n + 0 : (N - 1);
          int n1 = (n + 1 < N) ? n + 1 : (N - 1);
          int n2 = (n + 2 < N) ? n + 2 : (N - 1);
          int n3 = (n + 3 < N) ? n + 3 : (N - 1);
          bool kv = (k < K);
          f0 = (kv && n + 0 < N) ? Bm[kr * ldb + n0] : 0.0f;
          f1 = (kv && n + 1 < N) ? Bm[kr * ldb + n1] : 0.0f;
          f2 = (kv && n + 2 < N) ? Bm[kr * ldb + n2] : 0.0f;
          f3 = (kv && n + 3 < N) ? Bm[kr * ldb + n3] : 0.0f;
        }
        shBt[c + 0][r] = (_Float16)f0;
        shBt[c + 1][r] = (_Float16)f1;
        shBt[c + 2][r] = (_Float16)f2;
        shBt[c + 3][r] = (_Float16)f3;
      }
    }
    __syncthreads();

    // ---- two 16x16x32 WMMA sub-steps over the 64-wide K tile ----
    #pragma unroll
    for (int s = 0; s < 2; s++) {
      const int so = s * 32;
      v16h af[2], bf[2];
      #pragma unroll
      for (int i = 0; i < 2; i++) {
        int row = wm * 32 + i * 16 + lr;
        v8h lo = *(const v8h*)&shA[row][so + (lh << 3)];
        v8h hi = *(const v8h*)&shA[row][so + 16 + (lh << 3)];
        af[i] = __builtin_shufflevector(lo, hi, 0,1,2,3,4,5,6,7,8,9,10,11,12,13,14,15);
      }
      #pragma unroll
      for (int j = 0; j < 2; j++) {
        int col = wn * 32 + j * 16 + lr;
        v8h lo = *(const v8h*)&shBt[col][so + (lh << 3)];
        v8h hi = *(const v8h*)&shBt[col][so + 16 + (lh << 3)];
        bf[j] = __builtin_shufflevector(lo, hi, 0,1,2,3,4,5,6,7,8,9,10,11,12,13,14,15);
      }
      #pragma unroll
      for (int i = 0; i < 2; i++)
        #pragma unroll
        for (int j = 0; j < 2; j++)
          acc[i][j] = __builtin_amdgcn_wmma_f32_16x16x32_f16(
              false, af[i], false, bf[j], (short)0, acc[i][j], false, false);
    }
    __syncthreads();
  }

  // ---- epilogue ----
  #pragma unroll
  for (int i = 0; i < 2; i++) {
    #pragma unroll
    for (int j = 0; j < 2; j++) {
      int n = bn + wn * 32 + j * 16 + lr;
      #pragma unroll
      for (int r = 0; r < 8; r++) {
        int m = bm + wm * 32 + i * 16 + r + (lh << 3);
        if (m < M && n < N) {
          float v = alpha * acc[i][j][r];
          long co = (long)m * ldc + n;
          if (accumulate) {
            C[co] += v;
          } else {
            if (bias1) v += bias1[n];
            if (b2)    v += b2[(long)m * ldc + n];
            if (act == 1) v = gelu_erf(v);
            else if (act == 2) v = fmaxf(v, 0.0f);
            C[co] = v;
          }
        }
      }
    }
  }
}

// ---------------------------------------------------------------------------
// Elementwise / reduction helper kernels
// ---------------------------------------------------------------------------
__global__ void embed_kernel(const int* grid, const float* emb_w, const float* emb_b,
                             const float* pe, float* x)
{
  long i = (long)blockIdx.x * blockDim.x + threadIdx.x;
  if (i >= (long)B * S * D) return;
  int d  = (int)(i % D);
  long bs = i / D;
  int s  = (int)(bs % S);
  int c  = grid[bs];
  x[i] = emb_w[c * D + d] + emb_b[d] + pe[(long)s * D + d];
}

__global__ void geobias_kernel(const int* dist_idx, const int* dir_idx,
                               const float* dist_emb, const float* dir_emb, float* bias)
{
  long ij = (long)blockIdx.x * blockDim.x + threadIdx.x;
  if (ij >= (long)S * S) return;
  int di = dist_idx[ij], dr = dir_idx[ij];
  #pragma unroll
  for (int h = 0; h < NH; h++)
    bias[(long)h * S * S + ij] = dist_emb[di * NH + h] + dir_emb[dr * NH + h];
}

__global__ void softmax_kernel(float* x, int n)
{
  __shared__ float red[256];
  long row = blockIdx.x;
  float* p = x + row * (long)n;
  int tid = threadIdx.x;
  float mx = -3.4e38f;
  for (int i = tid; i < n; i += 256) mx = fmaxf(mx, p[i]);
  red[tid] = mx; __syncthreads();
  for (int s2 = 128; s2 > 0; s2 >>= 1) { if (tid < s2) red[tid] = fmaxf(red[tid], red[tid + s2]); __syncthreads(); }
  mx = red[0]; __syncthreads();
  float sum = 0.0f;
  for (int i = tid; i < n; i += 256) { float e = expf(p[i] - mx); p[i] = e; sum += e; }
  red[tid] = sum; __syncthreads();
  for (int s2 = 128; s2 > 0; s2 >>= 1) { if (tid < s2) red[tid] += red[tid + s2]; __syncthreads(); }
  float inv = 1.0f / red[0];
  for (int i = tid; i < n; i += 256) p[i] *= inv;
}

// out = LN(a + res) * g + b   (rows of length D; out may alias res)
__global__ void ln_residual_kernel(const float* a, const float* res,
                                   const float* g, const float* b, float* out)
{
  __shared__ float red[256];
  long row = blockIdx.x;
  const float* pa = a + row * D;
  const float* pr = res + row * D;
  float* po = out + row * D;
  int tid = threadIdx.x;
  bool has2 = (tid + 256) < D;
  float x0 = pa[tid] + pr[tid];
  float x1 = has2 ? (pa[tid + 256] + pr[tid + 256]) : 0.0f;
  red[tid] = x0 + x1; __syncthreads();
  for (int s2 = 128; s2 > 0; s2 >>= 1) { if (tid < s2) red[tid] += red[tid + s2]; __syncthreads(); }
  float mean = red[0] * (1.0f / D); __syncthreads();
  float d0 = x0 - mean, d1 = has2 ? (x1 - mean) : 0.0f;
  red[tid] = d0 * d0 + d1 * d1; __syncthreads();
  for (int s2 = 128; s2 > 0; s2 >>= 1) { if (tid < s2) red[tid] += red[tid + s2]; __syncthreads(); }
  float rstd = rsqrtf(red[0] * (1.0f / D) + LN_EPS);
  po[tid] = d0 * rstd * g[tid] + b[tid];
  if (has2) po[tid + 256] = d1 * rstd * g[tid + 256] + b[tid + 256];
}

__global__ void meanS_kernel(const float* x, float* g)
{
  int i = blockIdx.x * blockDim.x + threadIdx.x;
  if (i >= B * D) return;
  int b = i / D, d = i - b * D;
  const float* p = x + (long)b * S * D + d;
  float s = 0.0f;
  for (int k = 0; k < S; k++) s += p[(long)k * D];
  g[i] = s * (1.0f / S);
}

__global__ void tparams_kernel(const float* g,
    const float* w_rot, const float* b_rot, const float* w_refl, const float* b_refl,
    const float* w_tr, const float* b_tr, const float* w_sc, const float* b_sc,
    float* tp)
{
  __shared__ float lg[17];
  int b = blockIdx.x;
  int t = threadIdx.x;
  const float* gv = g + b * D;
  if (t < 17) {
    const float* w; const float* bb; int col, nc;
    if (t < 4)       { w = w_rot;  bb = b_rot;  col = t;      nc = 4; }
    else if (t < 12) { w = w_refl; bb = b_refl; col = t - 4;  nc = 8; }
    else if (t < 14) { w = w_tr;   bb = b_tr;   col = t - 12; nc = 2; }
    else             { w = w_sc;   bb = b_sc;   col = t - 14; nc = 3; }
    float s = bb[col];
    for (int d = 0; d < D; d++) s += gv[d] * w[d * nc + col];
    lg[t] = s;
  }
  __syncthreads();
  if (t == 0) {
    float* o = tp + b * 17;
    const int starts[3] = {0, 4, 14};
    const int lens[3]   = {4, 8, 3};
    for (int gi = 0; gi < 3; gi++) {
      int st = starts[gi], ln = lens[gi];
      float mx = -3.4e38f;
      for (int k = 0; k < ln; k++) mx = fmaxf(mx, lg[st + k]);
      float sm = 0.0f;
      for (int k = 0; k < ln; k++) { float e = expf(lg[st + k] - mx); lg[st + k] = e; sm += e; }
      float inv = 1.0f / sm;
      for (int k = 0; k < ln; k++) o[st + k] = lg[st + k] * inv;
    }
    o[12] = tanhf(lg[12]);
    o[13] = tanhf(lg[13]);
  }
}

// ti has padded row stride DT_PAD; cols [0,337) written, rest never read
__global__ void concat_kernel(const float* xs, const float* tp, float* ti)
{
  const int DT = D + 17;
  long i = (long)blockIdx.x * blockDim.x + threadIdx.x;
  if (i >= (long)B * S * DT) return;
  int j  = (int)(i % DT);
  long bs = i / DT;
  int b  = (int)(bs / S);
  ti[bs * DT_PAD + j] = (j < D) ? xs[bs * D + j] : tp[b * 17 + (j - D)];
}

__global__ void axpy_kernel(float* x, const float* t, float a, long n)
{
  long i = (long)blockIdx.x * blockDim.x + threadIdx.x;
  if (i < n) x[i] += a * t[i];
}

__global__ void relu_kernel(float* x, long n)
{
  long i = (long)blockIdx.x * blockDim.x + threadIdx.x;
  if (i < n) x[i] = fmaxf(x[i], 0.0f);
}

// ---------------------------------------------------------------------------
// Host orchestration
// ---------------------------------------------------------------------------
static inline int cdiv(long a, long b) { return (int)((a + b - 1) / b); }

static void launch_gemm(hipStream_t st, const float* A, const float* Bm,
                        const float* bias1, const float* bias2d, float* C,
                        int M, int N, int K, int lda, int ldb, int ldc,
                        long sA, long sA2, long sB, long sB2, long sC, long sC2,
                        long sBias2, int zInner, int Z, float alpha,
                        int transB, int act, int accum, int aMode, int dy, int dx)
{
  dim3 grd(cdiv(N, BN), cdiv(M, BM), Z);
  if (aMode)
    gemm_wmma_kernel<1, 0><<<grd, 256, 0, st>>>(A, Bm, bias1, bias2d, C, M, N, K,
        lda, ldb, ldc, sA, sA2, sB, sB2, sC, sC2, sBias2, zInner, alpha, act, accum, dy, dx);
  else if (transB)
    gemm_wmma_kernel<0, 1><<<grd, 256, 0, st>>>(A, Bm, bias1, bias2d, C, M, N, K,
        lda, ldb, ldc, sA, sA2, sB, sB2, sC, sC2, sBias2, zInner, alpha, act, accum, dy, dx);
  else
    gemm_wmma_kernel<0, 0><<<grd, 256, 0, st>>>(A, Bm, bias1, bias2d, C, M, N, K,
        lda, ldb, ldc, sA, sA2, sB, sB2, sC, sC2, sBias2, zInner, alpha, act, accum, dy, dx);
}

extern "C" void kernel_launch(void* const* d_in, const int* in_sizes, int n_in,
                              void* d_out, int out_size, void* d_ws, size_t ws_size,
                              hipStream_t stream) {
  (void)in_sizes; (void)n_in; (void)out_size; (void)ws_size;
  const int*   igrid    = (const int*)d_in[0];
  const float* pe       = (const float*)d_in[1];
  const float* in_emb_w = (const float*)d_in[2];
  const float* in_emb_b = (const float*)d_in[3];
  const float* wq = (const float*)d_in[4],  *bq = (const float*)d_in[5];
  const float* wk = (const float*)d_in[6],  *bk = (const float*)d_in[7];
  const float* wv = (const float*)d_in[8],  *bv = (const float*)d_in[9];
  const float* wo = (const float*)d_in[10], *bo = (const float*)d_in[11];
  const float* ln_a_g = (const float*)d_in[12], *ln_a_b = (const float*)d_in[13];
  const float* dist_emb = (const float*)d_in[14], *dir_emb = (const float*)d_in[15];
  const float* w_rot = (const float*)d_in[16], *b_rot = (const float*)d_in[17];
  const float* w_refl = (const float*)d_in[18], *b_refl = (const float*)d_in[19];
  const float* w_tr = (const float*)d_in[20], *b_tr = (const float*)d_in[21];
  const float* w_sc = (const float*)d_in[22], *b_sc = (const float*)d_in[23];
  const float* tn_w1 = (const float*)d_in[24], *tn_b1 = (const float*)d_in[25];
  const float* tn_w2 = (const float*)d_in[26], *tn_b2 = (const float*)d_in[27];
  const float* tn_w3 = (const float*)d_in[28], *tn_b3 = (const float*)d_in[29];
  const float* ff_w1 = (const float*)d_in[30], *ff_b1 = (const float*)d_in[31];
  const float* ff_w2 = (const float*)d_in[32], *ff_b2 = (const float*)d_in[33];
  const float* ln2_g = (const float*)d_in[34], *ln2_b = (const float*)d_in[35];
  const float* ckk[4] = {(const float*)d_in[36], (const float*)d_in[38],
                         (const float*)d_in[40], (const float*)d_in[42]};
  const float* cbb[4] = {(const float*)d_in[37], (const float*)d_in[39],
                         (const float*)d_in[41], (const float*)d_in[43]};
  const float* fus_w = (const float*)d_in[44], *fus_b = (const float*)d_in[45];
  const float* op_w1 = (const float*)d_in[46], *op_b1 = (const float*)d_in[47];
  const float* op_w2 = (const float*)d_in[48], *op_b2 = (const float*)d_in[49];
  const float* op_w3 = (const float*)d_in[50], *op_b3 = (const float*)d_in[51];
  const int* dist_idx = (const int*)d_in[52];
  const int* dir_idx  = (const int*)d_in[53];
  float* out = (float*)d_out;

  // workspace carve (floats)
  float* wsp = (float*)d_ws;
  auto carve = [&](long n) { float* r = wsp; wsp += (n + 255) & ~255L; return r; };
  float* buf_x   = carve((long)B * S * D);
  float* buf_q   = carve((long)B * S * D);
  float* buf_k   = carve((long)B * S * D);
  float* buf_v   = carve((long)B * S * D);
  float* buf_t1  = carve((long)B * S * 2 * D);   // attn-out / tn hidden
  float* buf_t2  = carve((long)B * S * D);
  float* buf_ff  = carve((long)B * S * FF);
  float* buf_ti  = carve((long)B * S * DT_PAD);
  float* buf_fe  = carve((long)B * S * 4 * D);
  float* buf_g   = carve((long)B * D);
  float* buf_tp  = carve(256);
  float* geo     = carve((long)NH * S * S);
  float* scores  = carve((long)B * NH * S * S);

  const int MBS = B * S;            // 3600
  const float scale = 1.0f / sqrtf((float)DK);

  // ---- embedding + positional encoding ----
  embed_kernel<<<cdiv((long)B * S * D, 256), 256, 0, stream>>>(igrid, in_emb_w, in_emb_b, pe, buf_x);

  for (int l = 0; l < L; l++) {
    const float* wql = wq + (long)l * D * D, *bql = bq + l * D;
    const float* wkl = wk + (long)l * D * D, *bkl = bk + l * D;
    const float* wvl = wv + (long)l * D * D, *bvl = bv + l * D;
    const float* wol = wo + (long)l * D * D, *bol = bo + l * D;

    // geometric bias for this layer: geo[NH,S,S]
    geobias_kernel<<<cdiv((long)S * S, 256), 256, 0, stream>>>(
        dist_idx, dir_idx, dist_emb + (long)l * 60 * NH, dir_emb + (long)l * 8 * NH, geo);

    // QKV projections: [B*S,D] @ [D,D] + b
    launch_gemm(stream, buf_x, wql, bql, 0, buf_q, MBS, D, D, D, D, D,
                0,0,0,0,0,0,0, 1, 1, 1.0f, 0, 0, 0, 0, 0, 0);
    launch_gemm(stream, buf_x, wkl, bkl, 0, buf_k, MBS, D, D, D, D, D,
                0,0,0,0,0,0,0, 1, 1, 1.0f, 0, 0, 0, 0, 0, 0);
    launch_gemm(stream, buf_x, wvl, bvl, 0, buf_v, MBS, D, D, D, D, D,
                0,0,0,0,0,0,0, 1, 1, 1.0f, 0, 0, 0, 0, 0, 0);

    // scores[z=b*NH+h] = scale * q_h @ k_h^T + geo[h]   (strided heads, no transposes)
    launch_gemm(stream, buf_q, buf_k, 0, geo, scores, S, S, DK, D, D, S,
                (long)S * D, DK, (long)S * D, DK,
                (long)NH * S * S, (long)S * S, (long)S * S,
                NH, B * NH, scale, /*transB=*/1, 0, 0, 0, 0, 0);

    softmax_kernel<<<B * NH * S, 256, 0, stream>>>(scores, S);

    // attn @ v -> buf_t1 as [B,S,D] (head column slices)
    launch_gemm(stream, scores, buf_v, 0, 0, buf_t1, S, DK, S, S, D, D,
                (long)NH * S * S, (long)S * S, (long)S * D, DK,
                (long)S * D, DK, 0, NH, B * NH, 1.0f, 0, 0, 0, 0, 0, 0);

    // output projection + residual LN
    launch_gemm(stream, buf_t1, wol, bol, 0, buf_t2, MBS, D, D, D, D, D,
                0,0,0,0,0,0,0, 1, 1, 1.0f, 0, 0, 0, 0, 0, 0);
    ln_residual_kernel<<<MBS, 256, 0, stream>>>(buf_t2, buf_x,
        ln_a_g + l * D, ln_a_b + l * D, buf_x);

    // geometric transform params
    meanS_kernel<<<cdiv(B * D, 256), 256, 0, stream>>>(buf_x, buf_g);
    tparams_kernel<<<B, 32, 0, stream>>>(buf_g,
        w_rot + (long)l * D * 4, b_rot + l * 4,
        w_refl + (long)l * D * 8, b_refl + l * 8,
        w_tr + (long)l * D * 2, b_tr + l * 2,
        w_sc + (long)l * D * 3, b_sc + l * 3, buf_tp);
    concat_kernel<<<cdiv((long)B * S * (D + 17), 256), 256, 0, stream>>>(buf_x, buf_tp, buf_ti);

    // transform network: (D+17)->2D gelu ->D gelu ->D  (A row stride padded to DT_PAD)
    launch_gemm(stream, buf_ti, tn_w1 + (long)l * (D + 17) * 2 * D, tn_b1 + l * 2 * D, 0,
                buf_ff, MBS, 2 * D, D + 17, DT_PAD, 2 * D, 2 * D,
                0,0,0,0,0,0,0, 1, 1, 1.0f, 0, 1, 0, 0, 0, 0);
    launch_gemm(stream, buf_ff, tn_w2 + (long)l * 2 * D * D, tn_b2 + l * D, 0,
                buf_t1, MBS, D, 2 * D, 2 * D, D, D,
                0,0,0,0,0,0,0, 1, 1, 1.0f, 0, 1, 0, 0, 0, 0);
    launch_gemm(stream, buf_t1, tn_w3 + (long)l * D * D, tn_b3 + l * D, 0,
                buf_t2, MBS, D, D, D, D, D,
                0,0,0,0,0,0,0, 1, 1, 1.0f, 0, 0, 0, 0, 0, 0);
    axpy_kernel<<<cdiv((long)B * S * D, 256), 256, 0, stream>>>(buf_x, buf_t2, 0.3f, (long)B * S * D);

    // feed-forward + residual LN
    launch_gemm(stream, buf_x, ff_w1 + (long)l * D * FF, ff_b1 + l * FF, 0,
                buf_ff, MBS, FF, D, D, FF, FF,
                0,0,0,0,0,0,0, 1, 1, 1.0f, 0, 1, 0, 0, 0, 0);
    launch_gemm(stream, buf_ff, ff_w2 + (long)l * FF * D, ff_b2 + l * D, 0,
                buf_t1, MBS, D, FF, FF, D, D,
                0,0,0,0,0,0,0, 1, 1, 1.0f, 0, 0, 0, 0, 0, 0);
    ln_residual_kernel<<<MBS, 256, 0, stream>>>(buf_t1, buf_x,
        ln2_g + l * D, ln2_b + l * D, buf_x);
  }

  // ---- multiscale conv fusion: each conv = k*k accumulating GEMM taps ----
  const int ksz[4] = {1, 3, 5, 7};
  for (int ci = 0; ci < 4; ci++) {
    int kk = ksz[ci], pad = kk / 2;
    int first = 1;
    for (int ky = 0; ky < kk; ky++) {
      for (int kx = 0; kx < kk; kx++) {
        const float* Wt = ckk[ci] + ((long)ky * kk + kx) * D * D;   // HWIO tap [D,D]
        launch_gemm(stream, buf_x, Wt, first ? cbb[ci] : 0, 0,
                    buf_fe + ci * D, MBS, D, D, D, D, 4 * D,
                    0,0,0,0,0,0,0, 1, 1, 1.0f, 0, 0, first ? 0 : 1,
                    /*aMode=*/1, ky - pad, kx - pad);
        first = 0;
      }
    }
  }
  relu_kernel<<<cdiv((long)B * S * 4 * D, 256), 256, 0, stream>>>(buf_fe, (long)B * S * 4 * D);

  // fusion + output head
  launch_gemm(stream, buf_fe, fus_w, fus_b, 0, buf_t1, MBS, D, 4 * D, 4 * D, D, D,
              0,0,0,0,0,0,0, 1, 1, 1.0f, 0, 0, 0, 0, 0, 0);
  launch_gemm(stream, buf_t1, op_w1, op_b1, 0, buf_q, MBS, D / 2, D, D, D / 2, D / 2,
              0,0,0,0,0,0,0, 1, 1, 1.0f, 0, 1, 0, 0, 0, 0);
  launch_gemm(stream, buf_q, op_w2, op_b2, 0, buf_k, MBS, D / 4, D / 2, D / 2, D / 4, D / 4,
              0,0,0,0,0,0,0, 1, 1, 1.0f, 0, 1, 0, 0, 0, 0);
  launch_gemm(stream, buf_k, op_w3, op_b3, 0, out, MBS, 10, D / 4, D / 4, 10, 10,
              0,0,0,0,0,0,0, 1, 1, 1.0f, 0, 0, 0, 0, 0, 0);
}
